// LlamaAttention_1065151889795
// MI455X (gfx1250) — compile-verified, matrix operand order checked by reading, not run on hardware
//
#include <hip/hip_runtime.h>
#include <hip/hip_bf16.h>
#include <math.h>

// ---------------- problem constants ----------------
#define B_    2
#define S_    2048
#define H_    4096
#define NH_   32
#define NKV_  8
#define HD_   128
#define REP_  (NH_ / NKV_)
#define SCALE_ 0.08838834764831845f   // 1/sqrt(128)
#define NEGBIG_ (-3.0e38f)

typedef __bf16 bf16_t;
typedef __attribute__((ext_vector_type(16))) __bf16 v16bf;
typedef __attribute__((ext_vector_type(8)))  __bf16 v8bf;
typedef __attribute__((ext_vector_type(8)))  float  v8f;
typedef __attribute__((ext_vector_type(4)))  float  v4f;

// ---------------- CDNA5 async global->LDS (ASYNCcnt-tracked DMA) ----------------
// GLOBAL_LOAD_ASYNC_TO_LDS_B128: per-lane, 16 bytes memory -> LDS, no VGPR data.
__device__ __forceinline__ void async_g2l_b128(unsigned lds_addr, const bf16_t* gptr) {
  asm volatile("global_load_async_to_lds_b128 %0, %1, off"
               :: "v"(lds_addr), "v"(gptr)
               : "memory");
}
__device__ __forceinline__ void wait_async() {
  asm volatile("s_wait_asynccnt 0x0" ::: "memory");
}
// flat->LDS byte offset: on CDNA5 the LDS aperture keeps the offset in addr[31:0]
__device__ __forceinline__ unsigned lds_off(const void* p) {
  return (unsigned)(unsigned long long)p;
}

// ---------------- WMMA helpers ----------------
__device__ __forceinline__ v8f wmma_bf16(v16bf a, v16bf b, v8f c) {
  // D = A(16x32 bf16) * B(32x16 bf16) + C(16x16 f32)
  return __builtin_amdgcn_wmma_f32_16x16x32_bf16(
      /*neg_a=*/false, a, /*neg_b=*/false, b,
      /*c_mod=*/(short)0, c, /*reuse_a=*/false, /*reuse_b=*/false);
}

// Load a 16x32 bf16 fragment from an LDS tile stored row-major with leading dim `ld`.
// Lane l<16 holds row (row0+l), K = {k0..k0+7, k0+16..k0+23}
// Lane l>=16 holds row (row0+l-16), K = {k0+8..k0+15, k0+24..k0+31}
__device__ __forceinline__ v16bf frag_ld(const bf16_t* base, int ld, int row0, int k0, int lane) {
  const int half = lane >> 4;
  const int r    = lane & 15;
  const bf16_t* p = base + (row0 + r) * ld + k0 + half * 8;
  v8bf lo = *(const v8bf*)(p);
  v8bf hi = *(const v8bf*)(p + 16);
  v16bf f;
#pragma unroll
  for (int i = 0; i < 8; ++i) { f[i] = lo[i]; f[i + 8] = hi[i]; }
  return f;
}

// ---------------- fp32 -> bf16 conversion ----------------
__global__ __launch_bounds__(256) void cvt_f32_bf16(const float* __restrict__ in,
                                                    bf16_t* __restrict__ out, long n4) {
  long i = (long)blockIdx.x * blockDim.x + threadIdx.x;
  if (i < n4) {
    v4f v = *(const v4f*)(in + i * 4);
    bf16_t* o = out + i * 4;
    o[0] = (bf16_t)v[0]; o[1] = (bf16_t)v[1];
    o[2] = (bf16_t)v[2]; o[3] = (bf16_t)v[3];
  }
}

// ---------------- generic GEMM: C[M,N] = A[M,K] * W[N,K]^T (bf16 in, f32 acc) ----------
// Block tile 64(M) x 128(N), K-step 64. 8 waves: 2(M) x 4(N), each wave 32x32.
// Double-buffered LDS (48 KB); tiles staged with GLOBAL_LOAD_ASYNC_TO_LDS_B128 so the
// DMA of tile k+1 overlaps the 8 WMMAs/wave on tile k; one barrier per 64-deep K step.
template <bool STORE_BF16>
__global__ __launch_bounds__(256) void gemm_xwT(const bf16_t* __restrict__ A,
                                                const bf16_t* __restrict__ W,
                                                void* __restrict__ Cv,
                                                int M, int N, int K) {
  __shared__ __attribute__((aligned(16))) bf16_t As[2][64 * 64];    // 2 x 8 KB
  __shared__ __attribute__((aligned(16))) bf16_t Bs[2][128 * 64];   // 2 x 16 KB
  const int tid  = threadIdx.x;
  const int wave = tid >> 5, lane = tid & 31;
  const int wm = wave & 1, wn = wave >> 1;
  const int m0 = blockIdx.y * 64, n0 = blockIdx.x * 128;

  v8f acc[2][2] = {};

  auto stage = [&](int buf, int k0) {
#pragma unroll
    for (int j = 0; j < 2; ++j) {          // A tile: 64 x 64 = 512 x 16B chunks
      const int it = tid + j * 256;
      const int row = it >> 3, c8 = (it & 7) * 8;
      async_g2l_b128(lds_off(&As[buf][row * 64 + c8]),
                     &A[(size_t)(m0 + row) * K + k0 + c8]);
    }
#pragma unroll
    for (int j = 0; j < 4; ++j) {          // B tile: 128 x 64 = 1024 x 16B chunks
      const int it = tid + j * 256;
      const int row = it >> 3, c8 = (it & 7) * 8;
      async_g2l_b128(lds_off(&Bs[buf][row * 64 + c8]),
                     &W[(size_t)(n0 + row) * K + k0 + c8]);
    }
  };

  stage(0, 0);
  int buf = 0;
  for (int k0 = 0; k0 < K; k0 += 64) {
    wait_async();
    __syncthreads();                 // current buf fully in LDS, prev reads retired
    if (k0 + 64 < K) stage(buf ^ 1, k0 + 64);
#pragma unroll
    for (int kk = 0; kk < 2; ++kk) {
#pragma unroll
      for (int mi = 0; mi < 2; ++mi) {
        v16bf af = frag_ld(As[buf], 64, wm * 32 + mi * 16, kk * 32, lane);
#pragma unroll
        for (int ni = 0; ni < 2; ++ni) {
          v16bf bfr = frag_ld(Bs[buf], 64, wn * 32 + ni * 16, kk * 32, lane);
          acc[mi][ni] = wmma_bf16(af, bfr, acc[mi][ni]);
        }
      }
    }
    buf ^= 1;
  }

  const int half = lane >> 4, r = lane & 15;
#pragma unroll
  for (int mi = 0; mi < 2; ++mi)
#pragma unroll
    for (int ni = 0; ni < 2; ++ni)
#pragma unroll
      for (int i = 0; i < 8; ++i) {
        const int row = m0 + wm * 32 + mi * 16 + i + 8 * half;
        const int col = n0 + wn * 32 + ni * 16 + r;
        if (STORE_BF16)
          ((bf16_t*)Cv)[(size_t)row * N + col] = (bf16_t)acc[mi][ni][i];
        else
          ((float*)Cv)[(size_t)row * N + col] = acc[mi][ni][i];
      }
}

// ---------------- RoPE (in place on bf16 [tok][nh*HD]) ----------------
__global__ __launch_bounds__(256) void rope_kernel(bf16_t* __restrict__ X,
                                                   const float* __restrict__ cosb,
                                                   const float* __restrict__ sinb,
                                                   int nh, long total) {
  long idx = (long)blockIdx.x * blockDim.x + threadIdx.x;  // over B*S*nh*64
  if (idx >= total) return;
  const int  d    = (int)(idx & 63);
  const long rest = idx >> 6;
  const int  h    = (int)(rest % nh);
  const long tok  = rest / nh;
  const float c1 = cosb[tok * HD_ + d],      s1 = sinb[tok * HD_ + d];
  const float c2 = cosb[tok * HD_ + d + 64], s2 = sinb[tok * HD_ + d + 64];
  bf16_t* p = X + (size_t)tok * nh * HD_ + (size_t)h * HD_;
  const float x1 = (float)p[d], x2 = (float)p[d + 64];
  p[d]      = (bf16_t)(x1 * c1 - x2 * s1);
  p[d + 64] = (bf16_t)(x2 * c2 + x1 * s2);
}

// ---------------- Pass A: raw masked scores + online row max/sum ----------------
// Block: 128 query rows x one (b,h). 8 waves, each owns 16 rows over all key tiles.
// K tiles double-buffered via async DMA: fetch of tile t+1 overlaps WMMA on tile t.
__global__ __launch_bounds__(256) void scores_kernel(const bf16_t* __restrict__ Qb,
                                                     const bf16_t* __restrict__ Kb,
                                                     float* __restrict__ attnW,
                                                     float* __restrict__ Ms,
                                                     float* __restrict__ Ls) {
  __shared__ __attribute__((aligned(16))) bf16_t Qs[128 * HD_];     // 32 KB
  __shared__ __attribute__((aligned(16))) bf16_t Ks[2][32 * HD_];   // 2 x 8 KB
  const int b = blockIdx.z, h = blockIdx.y, s0 = blockIdx.x * 128;
  const int g = h / REP_;
  const int tid = threadIdx.x, wave = tid >> 5, lane = tid & 31;
  const int half = lane >> 4, r = lane & 15;

  // stage Q tile: 128 rows x HD_ (lives for the whole block)
  for (int it = tid; it < 128 * 16; it += 256) {
    const int row = it >> 4, c8 = (it & 15) * 8;
    *(v8bf*)&Qs[row * HD_ + c8] =
        *(const v8bf*)&Qb[((size_t)(b * S_ + s0 + row)) * (NH_ * HD_) + h * HD_ + c8];
  }

  auto stageK = [&](int buf, int t0) {
#pragma unroll
    for (int j = 0; j < 2; ++j) {          // 32 x HD_ = 512 x 16B chunks
      const int it = tid + j * 256;
      const int row = it >> 4, c8 = (it & 15) * 8;
      async_g2l_b128(lds_off(&Ks[buf][row * HD_ + c8]),
                     &Kb[((size_t)(b * S_ + t0 + row)) * (NKV_ * HD_) + g * HD_ + c8]);
    }
  };

  float m_run[8], l_run[8];
#pragma unroll
  for (int i = 0; i < 8; ++i) { m_run[i] = NEGBIG_; l_run[i] = 0.0f; }

  const size_t rowbase = (size_t)(b * NH_ + h) * S_;
  const int ntiles = (s0 + 128) / 32;  // causal: only tiles that can contain valid entries

  stageK(0, 0);
  int buf = 0;
  for (int tt = 0; tt < ntiles; ++tt) {
    const int t0 = tt * 32;
    wait_async();
    __syncthreads();                 // K tile landed; prev-buffer reads retired
    if (tt + 1 < ntiles) stageK(buf ^ 1, t0 + 32);

    v8f acc[2] = {};
#pragma unroll
    for (int kk = 0; kk < 4; ++kk) {  // HD_/32
      v16bf af = frag_ld(Qs, HD_, wave * 16, kk * 32, lane);
#pragma unroll
      for (int ni = 0; ni < 2; ++ni) {
        v16bf bfr = frag_ld(Ks[buf], HD_, ni * 16, kk * 32, lane);
        acc[ni] = wmma_bf16(af, bfr, acc[ni]);
      }
    }

#pragma unroll
    for (int i = 0; i < 8; ++i) {
      const int row = s0 + wave * 16 + i + 8 * half;
      const int c0 = t0 + r, c1 = t0 + 16 + r;
      float v0 = acc[0][i] * SCALE_;
      float v1 = acc[1][i] * SCALE_;
      if (c0 > row) v0 = NEGBIG_;
      if (c1 > row) v1 = NEGBIG_;
      attnW[(rowbase + row) * S_ + c0] = v0;
      attnW[(rowbase + row) * S_ + c1] = v1;
      // online max / exp-sum across the 16 lanes of this half-wave
      float tmax = fmaxf(v0, v1);
#pragma unroll
      for (int off = 1; off < 16; off <<= 1) tmax = fmaxf(tmax, __shfl_xor(tmax, off));
      const float mnew = fmaxf(m_run[i], tmax);
      float e = __expf(v0 - mnew) + __expf(v1 - mnew);
#pragma unroll
      for (int off = 1; off < 16; off <<= 1) e += __shfl_xor(e, off);
      l_run[i] = l_run[i] * __expf(m_run[i] - mnew) + e;
      m_run[i] = mnew;
    }
    buf ^= 1;
  }

  if (r == 0) {
#pragma unroll
    for (int i = 0; i < 8; ++i) {
      const int srow = s0 + wave * 16 + i + 8 * half;
      Ms[(size_t)(b * NH_ + h) * S_ + srow] = m_run[i];
      Ls[(size_t)(b * NH_ + h) * S_ + srow] = l_run[i];
    }
  }
}

// ---------------- Pass B: normalize attn (write final weights) + attn @ V ----------------
// Block: 64 query rows x HD_ out cols for one (b,h). 8 waves 2(M)x4(N), wave tile 32x32.
__global__ __launch_bounds__(256) void av_kernel(const bf16_t* __restrict__ Vb,
                                                 const float* __restrict__ Ms,
                                                 const float* __restrict__ Ls,
                                                 float* __restrict__ attnW,
                                                 bf16_t* __restrict__ Obuf) {
  __shared__ __attribute__((aligned(16))) bf16_t At[64 * 32];    // normalized attn (4 KB)
  __shared__ __attribute__((aligned(16))) bf16_t Vt[HD_ * 32];   // V transposed [d][t] (8 KB)
  const int b = blockIdx.z, h = blockIdx.y, s0 = blockIdx.x * 64;
  const int g = h / REP_;
  const int tid = threadIdx.x, wave = tid >> 5, lane = tid & 31;
  const int wm = wave >> 2, wn = wave & 3;
  const size_t statbase = (size_t)(b * NH_ + h) * S_;
  const size_t rowbase  = statbase * S_;

  v8f acc[2][2] = {};
  const int full = S_ / 32;
  const int live = s0 / 32 + 2;  // tiles containing any valid (causal) entry

  for (int tt = 0; tt < full; ++tt) {
    const int t0 = tt * 32;
    if (tt < live) {
      __syncthreads();
      // normalize 64x32 raw scores in place; stage bf16 copy to LDS
      for (int it = tid; it < 64 * 4; it += 256) {
        const int row = it >> 2, c0 = (it & 3) * 8;
        const int srow = s0 + row;
        const float m  = Ms[statbase + srow];
        const float rl = 1.0f / Ls[statbase + srow];
        float* gp = &attnW[rowbase + (size_t)srow * S_ + t0 + c0];
#pragma unroll
        for (int j = 0; j < 8; ++j) {
          const float a = __expf(gp[j] - m) * rl;
          gp[j] = a;
          At[row * 32 + c0 + j] = (bf16_t)a;
        }
      }
      // stage V tile transposed: Vt[d][t]
      for (int it = tid; it < 32 * 8; it += 256) {
        const int trow = it >> 3, d0 = (it & 7) * 16;
        const bf16_t* vp = &Vb[((size_t)(b * S_ + t0 + trow)) * (NKV_ * HD_) + g * HD_ + d0];
#pragma unroll
        for (int j = 0; j < 16; ++j) Vt[(d0 + j) * 32 + trow] = vp[j];
      }
      __syncthreads();
#pragma unroll
      for (int mi = 0; mi < 2; ++mi) {
        v16bf af = frag_ld(At, 32, wm * 32 + mi * 16, 0, lane);
#pragma unroll
        for (int ni = 0; ni < 2; ++ni) {
          v16bf bfr = frag_ld(Vt, 32, wn * 32 + ni * 16, 0, lane);
          acc[mi][ni] = wmma_bf16(af, bfr, acc[mi][ni]);
        }
      }
    } else {
      // fully-masked tile: attention weights are exactly zero
      for (int it = tid; it < 64 * 4; it += 256) {
        const int row = it >> 2, c0 = (it & 3) * 8;
        float* gp = &attnW[rowbase + (size_t)(s0 + row) * S_ + t0 + c0];
#pragma unroll
        for (int j = 0; j < 8; ++j) gp[j] = 0.0f;
      }
    }
  }

  const int half = lane >> 4, r = lane & 15;
#pragma unroll
  for (int mi = 0; mi < 2; ++mi)
#pragma unroll
    for (int ni = 0; ni < 2; ++ni)
#pragma unroll
      for (int i = 0; i < 8; ++i) {
        const int srow = s0 + wm * 32 + mi * 16 + i + 8 * half;
        const int d    = wn * 32 + ni * 16 + r;
        Obuf[((size_t)(b * S_ + srow)) * (NH_ * HD_) + h * HD_ + d] = (bf16_t)acc[mi][ni][i];
      }
}

// ---------------- host: orchestrate pipeline ----------------
extern "C" void kernel_launch(void* const* d_in, const int* in_sizes, int n_in,
                              void* d_out, int out_size, void* d_ws, size_t ws_size,
                              hipStream_t stream) {
  (void)in_sizes; (void)n_in; (void)out_size; (void)ws_size;

  const float* hs   = (const float*)d_in[0];
  const float* cosb = (const float*)d_in[1];
  const float* sinb = (const float*)d_in[2];
  // d_in[3] attention_mask: causal mask is computed analytically in-kernel
  const float* Wq = (const float*)d_in[4];
  const float* Wk = (const float*)d_in[5];
  const float* Wv = (const float*)d_in[6];
  const float* Wo = (const float*)d_in[7];

  float* outp  = (float*)d_out;                       // (B,S,H) attn_output
  float* attnW = outp + (size_t)B_ * S_ * H_;         // (B,NH,S,S) attn_weights

  const size_t BS = (size_t)B_ * S_;  // 4096 tokens

  // workspace carve-out
  char* p = (char*)d_ws;
  auto carve = [&](size_t bytes) {
    void* r = (void*)p;
    p += (bytes + 255) & ~(size_t)255;
    return r;
  };
  bf16_t* Xb   = (bf16_t*)carve(BS * H_ * 2);                 // hidden bf16
  bf16_t* Wqb  = (bf16_t*)carve((size_t)NH_ * HD_ * H_ * 2);
  bf16_t* Wkb  = (bf16_t*)carve((size_t)NKV_ * HD_ * H_ * 2);
  bf16_t* Wvb  = (bf16_t*)carve((size_t)NKV_ * HD_ * H_ * 2);
  bf16_t* Wob  = (bf16_t*)carve((size_t)H_ * NH_ * HD_ * 2);
  bf16_t* Qb   = (bf16_t*)carve(BS * NH_ * HD_ * 2);
  bf16_t* Kb   = (bf16_t*)carve(BS * NKV_ * HD_ * 2);
  bf16_t* Vb   = (bf16_t*)carve(BS * NKV_ * HD_ * 2);
  bf16_t* Obuf = (bf16_t*)carve(BS * NH_ * HD_ * 2);
  float*  Msb  = (float*)carve((size_t)B_ * NH_ * S_ * 4);
  float*  Lsb  = (float*)carve((size_t)B_ * NH_ * S_ * 4);

  auto cvt = [&](const float* in, bf16_t* out, size_t n) {
    const long n4 = (long)(n / 4);
    const int blocks = (int)((n4 + 255) / 256);
    cvt_f32_bf16<<<blocks, 256, 0, stream>>>(in, out, n4);
  };

  // 1) down-convert activations + weights to bf16
  cvt(hs, Xb, BS * H_);
  cvt(Wq, Wqb, (size_t)NH_ * HD_ * H_);
  cvt(Wk, Wkb, (size_t)NKV_ * HD_ * H_);
  cvt(Wv, Wvb, (size_t)NKV_ * HD_ * H_);
  cvt(Wo, Wob, (size_t)H_ * NH_ * HD_);

  // 2) QKV projections: [BS,H] @ [N,H]^T
  gemm_xwT<true><<<dim3((NH_ * HD_) / 128, (int)BS / 64), 256, 0, stream>>>(
      Xb, Wqb, Qb, (int)BS, NH_ * HD_, H_);
  gemm_xwT<true><<<dim3((NKV_ * HD_) / 128, (int)BS / 64), 256, 0, stream>>>(
      Xb, Wkb, Kb, (int)BS, NKV_ * HD_, H_);
  gemm_xwT<true><<<dim3((NKV_ * HD_) / 128, (int)BS / 64), 256, 0, stream>>>(
      Xb, Wvb, Vb, (int)BS, NKV_ * HD_, H_);

  // 3) RoPE on Q and K (in place)
  {
    const long tq = (long)BS * NH_ * 64;
    rope_kernel<<<(int)((tq + 255) / 256), 256, 0, stream>>>(Qb, cosb, sinb, NH_, tq);
    const long tk = (long)BS * NKV_ * 64;
    rope_kernel<<<(int)((tk + 255) / 256), 256, 0, stream>>>(Kb, cosb, sinb, NKV_, tk);
  }

  // 4) Pass A: raw masked scores + row max/sum
  scores_kernel<<<dim3(S_ / 128, NH_, B_), 256, 0, stream>>>(Qb, Kb, attnW, Msb, Lsb);

  // 5) Pass B: normalize (final attn_weights) + attn @ V
  av_kernel<<<dim3(S_ / 64, NH_, B_), 256, 0, stream>>>(Vb, Msb, Lsb, attnW, Obuf);

  // 6) output projection: [BS, NH*HD] @ [H, NH*HD]^T -> fp32
  gemm_xwT<false><<<dim3(H_ / 128, (int)BS / 64), 256, 0, stream>>>(
      Obuf, Wob, outp, (int)BS, H_, NH_ * HD_);
}
